// policy_network_65343632442044
// MI455X (gfx1250) — matrix operand order repaired and checked by reading, not compiled, not run on hardware
//
#include <hip/hip_runtime.h>

// ---------------- problem constants ----------------
#define B_   128
#define T_   256
#define U_   512
#define G8   8        // gates
#define NB   32       // persistent blocks (one per u-tile of 16 columns)
#define NT   256      // threads per block = 8 waves (one per b-tile of 16 rows)

typedef __attribute__((ext_vector_type(16))) __bf16        v16bf;
typedef __attribute__((ext_vector_type(8)))  float         v8f;
typedef __attribute__((ext_vector_type(4)))  unsigned int  u32x4;
typedef __attribute__((ext_vector_type(4)))  float         f32x4;

// exact parameter type of the async-LDS builtin (from the round-2 diagnostic):
// __attribute__((vector_size(16))) int, in addrspace(1)/(3)
typedef int i32x4v __attribute__((vector_size(16)));
typedef __attribute__((address_space(1))) i32x4v* as1_v4i_p;
typedef __attribute__((address_space(3))) i32x4v* as3_v4i_p;

// ---------------- feature probes ----------------
#if __has_builtin(__builtin_amdgcn_global_load_async_to_lds_b128) && \
    __has_builtin(__builtin_amdgcn_s_wait_asynccnt)
#define HAVE_ASYNC_LDS 1
#else
#define HAVE_ASYNC_LDS 0
#endif

#if __has_builtin(__builtin_amdgcn_tanhf)
#define TANH_NATIVE 1
#else
#define TANH_NATIVE 0
#endif

// ---------------- workspace layout (bytes) ----------------
#define BAR_OFF   0
#define M_OFF     1024                       // double-buffered bf16 m: 2*128*512*2 = 256KB
#define M_BYTES   (2 * B_ * U_ * 2)
#define XT_OFF    (M_OFF + M_BYTES)          // transposed x: 256*128*4 = 128KB
#define XT_BYTES  (T_ * B_ * 4)
#define RT_OFF    (XT_OFF + XT_BYTES)        // packed bf16 weights: 4MB
#define RT_ELEMS  (U_ * 8 * U_)              // 512*4096 = 2,097,152 bf16

// ---------------- helpers ----------------
__device__ __forceinline__ unsigned short bf16b(float f) {
  unsigned u = __float_as_uint(f);
  return (unsigned short)((u + 0x7FFFu + ((u >> 16) & 1u)) >> 16);
}
__device__ __forceinline__ float sigf(float x) {
  return __fdividef(1.0f, 1.0f + __expf(-x));
}
__device__ __forceinline__ float tanf_(float x) {
#if TANH_NATIVE
  return __builtin_amdgcn_tanhf(x);          // v_tanh_f32 (gfx1250 TRANS op)
#else
  // 1 - 2/(e^{2x}+1): saturates correctly at +/-1 without NaNs
  return 1.0f - __fdividef(2.0f, __expf(2.0f * x) + 1.0f);
#endif
}

// ---------------- prep kernels ----------------
__global__ void prep_xt(const float* __restrict__ x, float* __restrict__ xT) {
  int idx = blockIdx.x * blockDim.x + threadIdx.x;   // idx = t*128 + b
  if (idx < B_ * T_) {
    int t = idx >> 7, b = idx & 127;
    xT[idx] = x[b * T_ + t];
  }
}

// Pack recurrent_kernel [512, 4096] fp32 (row-major, col = g*512+u) into the exact
// per-lane B-fragment order used here (k = 16h + e), bf16. Tile = (u_tile, gate, kb): 32 lanes x 16 bf16.
// Consumer mapping: lane<16 -> col = ut*16+lane,   K = kb*32 + e
//                   lane>=16 -> col = ut*16+lane-16, K = kb*32 + 16 + e
// Producer indexed so reads of R are fully coalesced (scatter the stores instead).
__global__ void prep_rt(const float* __restrict__ R, unsigned short* __restrict__ Rt) {
  int idx = blockIdx.x * blockDim.x + threadIdx.x;   // idx = k*4096 + col
  if (idx < RT_ELEMS) {
    int col = idx & 4095;
    int k   = idx >> 12;
    int g   = col >> 9;          // gate
    int n   = col & 511;         // u
    int ut  = n >> 4;            // u-tile
    int nl  = n & 15;            // lane-low
    int kb  = k >> 5;            // k-block of 32
    int kl  = k & 31;            // k within block
    int lane = (kl < 16) ? nl : (16 + nl);
    int e    = kl & 15;
    Rt[(((ut * 8 + g) * 16 + kb) << 9) + lane * 16 + e] = bf16b(R[idx]);
  }
}

__global__ void prep_init(unsigned int* __restrict__ mz, unsigned int* __restrict__ bar) {
  int idx = blockIdx.x * blockDim.x + threadIdx.x;
  if (idx < (M_BYTES / 4)) mz[idx] = 0u;           // zero both m buffers
  if (idx == 0) { bar[0] = 0u; bar[1] = 0u; }      // barrier count + generation
}

// ---------------- persistent NAS-RNN kernel ----------------
__global__ __launch_bounds__(NT) void nas_persistent(
    const float* __restrict__ xT,            // [T][B]
    const float* __restrict__ kernelW,       // [8*U]
    const unsigned short* __restrict__ Rt,   // packed bf16 B fragments
    unsigned short* __restrict__ mbuf,       // 2 x [B][U] bf16, double buffered
    unsigned int* __restrict__ bar,          // {count, gen}
    float* __restrict__ out)                 // [B][U]
{
  __shared__ unsigned short Bs[65536];       // 128KB: this u-tile's B slice (8 gates x K=512)

  const int tid  = threadIdx.x;
  const int wave = tid >> 5;                 // b-tile 0..7
  const int lane = tid & 31;
  const int l16  = lane & 15;
  const int hi   = (lane >> 4) & 1;
  const int ut   = blockIdx.x;               // u-tile 0..31
  const int u_abs = ut * 16 + l16;           // N column owned by this lane (C/D layout)

  // ---- stage B tiles into LDS (once; weights are step-invariant)
  {
    const unsigned short* src = Rt + (size_t)ut * 65536;
#if HAVE_ASYNC_LDS
    for (int i = tid; i < 65536 / 8; i += NT) {
      __builtin_amdgcn_global_load_async_to_lds_b128(
          (as1_v4i_p)(unsigned short*)(src + i * 8),
          (as3_v4i_p)(Bs + i * 8), 0, 0);
    }
    __builtin_amdgcn_s_wait_asynccnt(0);
#else
    const u32x4* s4 = (const u32x4*)src;
    u32x4* d4 = (u32x4*)Bs;
    for (int i = tid; i < 65536 / 8; i += NT) d4[i] = s4[i];
#endif
  }
  __syncthreads();

  // input-kernel scalars: fixed per lane (N is fixed per lane in C/D layout)
  float ker[G8];
#pragma unroll
  for (int g = 0; g < G8; ++g) ker[g] = kernelW[g * U_ + u_abs];

  float c[8];
#pragma unroll
  for (int j = 0; j < 8; ++j) c[j] = 0.f;
  float mn[8];
#pragma unroll
  for (int j = 0; j < 8; ++j) mn[j] = 0.f;

  const int rowb = wave * 16 + l16;          // A-fragment row for this lane
  const int o    = hi ? 16 : 0;              // A-fragment K sub-offset
  const int bofs = hi ? 8 : 0;               // C/D row-group base (M = bofs + j)

  for (int t = 0; t < T_; ++t) {
    const unsigned short* mr = mbuf + (t & 1) * (B_ * U_);
    unsigned short*       mw = mbuf + ((t + 1) & 1) * (B_ * U_);

    // x fragment: 8 contiguous floats of xT[t] for this lane's 8 M rows
    f32x4 xv0 = *(const f32x4*)(xT + t * B_ + wave * 16 + bofs);
    f32x4 xv1 = *(const f32x4*)(xT + t * B_ + wave * 16 + bofs + 4);

    v8f acc[G8] = {};

    const unsigned short* arow = mr + rowb * U_;
    for (int kb = 0; kb < 16; ++kb) {
      union { v16bf v; u32x4 q[2]; } A;
      A.q[0] = *(const u32x4*)(arow + kb * 32 + o);
      A.q[1] = *(const u32x4*)(arow + kb * 32 + o + 8);
#pragma unroll
      for (int g = 0; g < G8; ++g) {
        union { v16bf v; u32x4 q[2]; } Bf;
        const u32x4* bp = (const u32x4*)(Bs + ((g * 16 + kb) * 512 + lane * 16));
        Bf.q[0] = bp[0];
        Bf.q[1] = bp[1];
        acc[g] = __builtin_amdgcn_wmma_f32_16x16x32_bf16(
            false, A.v, false, Bf.v, (short)0, acc[g], false, false);
      }
    }

    // ---- fused NAS cell elementwise (8 rows per lane, C/D layout)
#pragma unroll
    for (int j = 0; j < 8; ++j) {
      float xb = (j < 4) ? xv0[j] : xv1[j - 4];
      float r0 = acc[0][j], r1 = acc[1][j], r2 = acc[2][j], r3 = acc[3][j];
      float r4 = acc[4][j], r5 = acc[5][j], r6 = acc[6][j], r7 = acc[7][j];
      float i0 = xb * ker[0], i1 = xb * ker[1], i2 = xb * ker[2], i3 = xb * ker[3];
      float i4 = xb * ker[4], i5 = xb * ker[5], i6 = xb * ker[6], i7 = xb * ker[7];

      float l10 = sigf(i0 + r0);
      float l11 = fmaxf(i1 + r1, 0.f);
      float l12 = sigf(i2 + r2);
      float l13 = fmaxf(i3 * r3, 0.f);
      float l14 = tanf_(i4 + r4);
      float l15 = sigf(i5 + r5);
      float l16g = tanf_(i6 + r6);
      float l17 = sigf(i7 + r7);

      float l20 = tanf_(l10 * l11);
      float l21 = tanf_(l12 + l13);
      float l22 = tanf_(l14 * l15);
      float l23 = sigf(l16g + l17);

      l20 = tanf_(l20 + c[j]);
      float nc  = l20 * l21;
      float l31 = tanf_(l22 + l23);
      float nm  = tanf_(nc * l31);

      c[j]  = nc;
      mn[j] = nm;
      mw[(wave * 16 + bofs + j) * U_ + u_abs] = bf16b(nm);
    }

    // ---- device-wide barrier between steps (skip after the last step)
    if (t != T_ - 1) {
      __syncthreads();
      __threadfence();                       // release: make m(t+1) stores visible at agent scope
      if (tid == 0) {
        unsigned g0 = __hip_atomic_load(&bar[1], __ATOMIC_RELAXED, __HIP_MEMORY_SCOPE_AGENT);
        unsigned a  = __hip_atomic_fetch_add(&bar[0], 1u, __ATOMIC_ACQ_REL, __HIP_MEMORY_SCOPE_AGENT);
        if (a == (unsigned)(NB - 1)) {
          __hip_atomic_store(&bar[0], 0u, __ATOMIC_RELAXED, __HIP_MEMORY_SCOPE_AGENT);
          __hip_atomic_fetch_add(&bar[1], 1u, __ATOMIC_RELEASE, __HIP_MEMORY_SCOPE_AGENT);
        } else {
          while (__hip_atomic_load(&bar[1], __ATOMIC_RELAXED, __HIP_MEMORY_SCOPE_AGENT) == g0)
            __builtin_amdgcn_s_sleep(1);
        }
      }
      __syncthreads();
      __threadfence();                       // acquire: invalidate L0 before reading new m
    }
  }

  // final output: last m + 0.05
#pragma unroll
  for (int j = 0; j < 8; ++j)
    out[(wave * 16 + bofs + j) * U_ + u_abs] = mn[j] + 0.05f;
}

// ---------------- launch ----------------
extern "C" void kernel_launch(void* const* d_in, const int* in_sizes, int n_in,
                              void* d_out, int out_size, void* d_ws, size_t ws_size,
                              hipStream_t stream) {
  const float* x  = (const float*)d_in[0];   // [128, 256]
  const float* kw = (const float*)d_in[1];   // [1, 4096]
  const float* rk = (const float*)d_in[2];   // [512, 4096]

  char* ws = (char*)d_ws;
  unsigned int*   bar  = (unsigned int*)(ws + BAR_OFF);
  unsigned short* mbuf = (unsigned short*)(ws + M_OFF);
  float*          xT   = (float*)(ws + XT_OFF);
  unsigned short* Rt   = (unsigned short*)(ws + RT_OFF);

  prep_init<<<(M_BYTES / 4 + 255) / 256, 256, 0, stream>>>((unsigned int*)mbuf, bar);
  prep_xt<<<(B_ * T_ + 255) / 256, 256, 0, stream>>>(x, xT);
  prep_rt<<<(RT_ELEMS + 255) / 256, 256, 0, stream>>>(rk, Rt);
  nas_persistent<<<NB, NT, 0, stream>>>(xT, kw, Rt, mbuf, bar, (float*)d_out);
}
